// WeLMMoeAttention_31576599560863
// MI455X (gfx1250) — matrix operand order repaired and checked by reading, not compiled, not run on hardware
//
#include <hip/hip_runtime.h>
#include <hip/hip_bf16.h>

// ---------------------------------------------------------------------------
// Problem constants (B=1)
// ---------------------------------------------------------------------------
#define SEQ     2048
#define HID     4096
#define NHEADS  32
#define NKVH    8
#define DHEAD   128
#define QSIZE   (NHEADS * DHEAD)          // 4096
#define KVSIZE  (NKVH * DHEAD)            // 1024
#define QKV_LD  (QSIZE + 2 * KVSIZE)      // 6144

typedef __bf16 bf16_t;
typedef __attribute__((ext_vector_type(16))) bf16_t v16bf;
typedef __attribute__((ext_vector_type(8)))  bf16_t v8bf;
typedef __attribute__((ext_vector_type(8)))  float  v8f;
typedef int v4i __attribute__((vector_size(16)));   // matches builtin prototype

#define AS1 __attribute__((address_space(1)))
#define AS3 __attribute__((address_space(3)))

// gfx1250 async global->LDS copy (ASYNCcnt tracked), with safe fallback.
#if defined(__AMDGCN__) && __has_builtin(__builtin_amdgcn_global_load_async_to_lds_b128)
#define HAVE_ASYNC_LDS 1
#else
#define HAVE_ASYNC_LDS 0
#endif

static __device__ __forceinline__ void async_copy_b128(const bf16_t* g, bf16_t* l) {
#if HAVE_ASYNC_LDS
  __builtin_amdgcn_global_load_async_to_lds_b128((AS1 v4i*)g, (AS3 v4i*)l, 0, 0);
#else
  *(v8bf*)l = *(const v8bf*)g;
#endif
}

static __device__ __forceinline__ void wait_async0() {
#if defined(__AMDGCN__)
#if __has_builtin(__builtin_amdgcn_s_wait_asynccnt)
  __builtin_amdgcn_s_wait_asynccnt(0);
#else
  asm volatile("s_wait_asynccnt 0" ::: "memory");
#endif
#endif
}

// ---------------------------------------------------------------------------
// WMMA helpers (CDNA5 V_WMMA_F32_16X16X32_BF16, D = A(16x32)*B(32x16) + C)
// ---------------------------------------------------------------------------
static __device__ __forceinline__ v8f wmma_bf16(v16bf a, v16bf b, v8f c) {
  // (neg_a, A, neg_b, B, c_mod, C, reuse_a, reuse_b)
  return __builtin_amdgcn_wmma_f32_16x16x32_bf16(false, a, false, b,
                                                 (short)0, c, false, false);
}

// A-fragment (16x32, 16-bit): lane = M row (lane&15); for lane<16 elements are
// K = 0..7 then 16..23; for lane>=16 K = 8..15 then 24..31.  kb8 = (lane>>4)*8.
static __device__ __forceinline__ v16bf lds_frag_a(const bf16_t* rowptr, int kb8) {
  v8bf lo = *(const v8bf*)(rowptr + kb8);
  v8bf hi = *(const v8bf*)(rowptr + kb8 + 16);
  v16bf r;
#pragma unroll
  for (int i = 0; i < 8; ++i) { r[i] = lo[i]; r[i + 8] = hi[i]; }
  return r;
}

// B-fragment (32x16, 16-bit): lane = N col (lane&15); for lane<16 elements are
// K = 0..7 then 16..23; for lane>=16 K = 8..15 then 24..31.  kh16 = (lane>>4)*16.
static __device__ __forceinline__ v16bf lds_frag_b(const bf16_t* rowptr, int kh16) {
  v8bf lo = *(const v8bf*)(rowptr + (kh16 >> 1));
  v8bf hi = *(const v8bf*)(rowptr + (kh16 >> 1) + 16);
  v16bf r;
#pragma unroll
  for (int i = 0; i < 8; ++i) { r[i] = lo[i]; r[i + 8] = hi[i]; }
  return r;
}

// ---------------------------------------------------------------------------
// Kernel 1&4: NT GEMM  C[M,N] = A[M,K] * W[N,K]^T (+ bias[N])
// A is fp32 or bf16 (template), W/bias fp32, C fp32 or bf16 (template).
// bf16 WMMA, f32 accumulation.  M,N % 128 == 0, K % 32 == 0.
// Block: 256 threads (8 waves), 128x128 tile, BK=32.  Wave tile 64x32.
// ---------------------------------------------------------------------------
template <bool A_IS_BF16, bool OUT_BF16>
__global__ __launch_bounds__(256)
void gemm_nt(const void* __restrict__ Araw, const float* __restrict__ W,
             const float* __restrict__ bias, void* __restrict__ Craw,
             int M, int N, int K) {
  __shared__ __align__(16) bf16_t lA[128 * 32];
  __shared__ __align__(16) bf16_t lB[128 * 32];

  const int tid   = threadIdx.x;
  const int lane  = tid & 31;
  const int wave  = tid >> 5;
  const int wm    = wave >> 2;            // 0..1 (M direction)
  const int wn    = wave & 3;             // 0..3 (N direction)
  const int l16   = lane & 15;
  const int kb8   = (lane >> 4) << 3;     // 0 or 8  (A frag)
  const int kh16  = (lane >> 4) << 4;     // 0 or 16 (B frag)
  const int half8 = (lane >> 4) << 3;     // row offset in C/D layout
  const int row0  = blockIdx.y * 128;
  const int col0  = blockIdx.x * 128;

  v8f zero = {};
  v8f acc[4][2];
#pragma unroll
  for (int i = 0; i < 4; ++i)
#pragma unroll
    for (int j = 0; j < 2; ++j) acc[i][j] = zero;

  const int trf = tid >> 3;               // fp32 staging: 0..31 row
  const int tcf = (tid & 7) << 2;         // fp32 staging: col (float4)
  const int trb = tid >> 2;               // bf16 staging: 0..63 row
  const int tcb = (tid & 3) << 3;         // bf16 staging: col (v8bf)

  for (int kb = 0; kb < K; kb += 32) {
    // ---- stage A(128x32) to LDS as bf16
    if constexpr (A_IS_BF16) {
      const bf16_t* A = (const bf16_t*)Araw;
#pragma unroll
      for (int p = 0; p < 2; ++p) {
        int r = p * 64 + trb;
        *(v8bf*)&lA[r * 32 + tcb] =
            *(const v8bf*)(A + (size_t)(row0 + r) * K + kb + tcb);
      }
    } else {
      const float* A = (const float*)Araw;
#pragma unroll
      for (int p = 0; p < 4; ++p) {
        int r = p * 32 + trf;
        float4 av = *(const float4*)(A + (size_t)(row0 + r) * K + kb + tcf);
        bf16_t* da = &lA[r * 32 + tcf];
        da[0] = (bf16_t)av.x; da[1] = (bf16_t)av.y;
        da[2] = (bf16_t)av.z; da[3] = (bf16_t)av.w;
      }
    }
    // ---- stage W(128x32) to LDS as bf16 (always fp32 source)
#pragma unroll
    for (int p = 0; p < 4; ++p) {
      int r = p * 32 + trf;
      float4 bv = *(const float4*)(W + (size_t)(col0 + r) * K + kb + tcf);
      bf16_t* db = &lB[r * 32 + tcf];
      db[0] = (bf16_t)bv.x; db[1] = (bf16_t)bv.y;
      db[2] = (bf16_t)bv.z; db[3] = (bf16_t)bv.w;
    }
    __syncthreads();

    v16bf aF[4], bF[2];
#pragma unroll
    for (int mi = 0; mi < 4; ++mi)
      aF[mi] = lds_frag_a(&lA[(wm * 64 + mi * 16 + l16) * 32], kb8);
#pragma unroll
    for (int ni = 0; ni < 2; ++ni)
      bF[ni] = lds_frag_b(&lB[(wn * 32 + ni * 16 + l16) * 32], kh16);
#pragma unroll
    for (int mi = 0; mi < 4; ++mi)
#pragma unroll
      for (int ni = 0; ni < 2; ++ni)
        acc[mi][ni] = wmma_bf16(aF[mi], bF[ni], acc[mi][ni]);
    __syncthreads();
  }

  // ---- epilogue: C/D layout -> global (+bias)
#pragma unroll
  for (int mi = 0; mi < 4; ++mi) {
#pragma unroll
    for (int ni = 0; ni < 2; ++ni) {
      int col = col0 + wn * 32 + ni * 16 + l16;
      float bv = bias ? bias[col] : 0.0f;
#pragma unroll
      for (int r = 0; r < 8; ++r) {
        int row = row0 + wm * 64 + mi * 16 + r + half8;
        float v = acc[mi][ni][r] + bv;
        if constexpr (OUT_BF16)
          ((bf16_t*)Craw)[(size_t)row * N + col] = (bf16_t)v;
        else
          ((float*)Craw)[(size_t)row * N + col] = v;
      }
    }
  }
}

// ---------------------------------------------------------------------------
// Kernel 2: in-place RoPE on q (32 heads) and k (8 kv heads), bf16 storage.
// One thread handles one (s, head, j) rotation pair (j in 0..63).
// ---------------------------------------------------------------------------
__global__ __launch_bounds__(256)
void rope_kernel(bf16_t* __restrict__ qkv, const int* __restrict__ positions) {
  int idx = blockIdx.x * blockDim.x + threadIdx.x;
  const int TOT = SEQ * (NHEADS + NKVH) * 64;
  if (idx >= TOT) return;
  int j  = idx & 63;
  int hh = (idx >> 6) % (NHEADS + NKVH);
  int s  = idx / (64 * (NHEADS + NKVH));
  bf16_t* base = qkv + (size_t)s * QKV_LD +
                 ((hh < NHEADS) ? hh * DHEAD : QSIZE + (hh - NHEADS) * DHEAD);
  float pos = (float)positions[s];
  // inv_freq = 10000^(-j/64) = exp(-j * ln(10000)/64)
  float inv = __expf(-(float)j * (9.210340371976184f / 64.0f));
  float ang = pos * inv;
  float c = __cosf(ang), sn = __sinf(ang);
  float x1 = (float)base[j], x2 = (float)base[j + 64];
  base[j]      = (bf16_t)(x1 * c - x2 * sn);
  base[j + 64] = (bf16_t)(x2 * c + x1 * sn);
}

// ---------------------------------------------------------------------------
// Kernel 3: causal flash attention with GQA (4 q-heads share a kv-head).
// Grid: (S/64 query blocks, NH heads).  Block: 128 threads (4 waves).
// Wave owns 16 query rows; loop over 32-key tiles with online softmax.
// Q/K tiles staged global->LDS via async b128 copies (bf16 end to end).
// Q*K^T and P*V both via bf16 WMMA (f32 accum).
// ---------------------------------------------------------------------------
__global__ __launch_bounds__(128)
void flash_attn(const bf16_t* __restrict__ qkv, bf16_t* __restrict__ attn) {
  __shared__ __align__(16) bf16_t lQ[64 * 128];       // query tile
  __shared__ __align__(16) bf16_t lK[32 * 128];       // key tile (key-major)
  __shared__ __align__(16) bf16_t lVt[128 * 32];      // V^T tile (dim-major)
  __shared__ __align__(16) bf16_t lP[4 * 16 * 32];    // per-wave probs

  const int tid   = threadIdx.x;
  const int lane  = tid & 31;
  const int wave  = tid >> 5;               // 0..3
  const int l16   = lane & 15;
  const int kb8   = (lane >> 4) << 3;
  const int kh16  = (lane >> 4) << 4;
  const int half8 = kb8;
  const int h     = blockIdx.y;
  const int qb    = blockIdx.x;
  const int qbase = qb * 64;
  const int kvh   = h >> 2;                 // NH/NKV = 4
  const float scale = 0.08838834764831845f; // 1/sqrt(128)

  const int trq = tid >> 4;                 // 0..7 staging row
  const int tcq = (tid & 15) << 3;          // 0..120 staging col (v8bf)

  // ---- stage Q (64x128 bf16): pure copy -> async global->LDS
#pragma unroll
  for (int p = 0; p < 8; ++p) {
    int r = p * 8 + trq;
    async_copy_b128(qkv + (size_t)(qbase + r) * QKV_LD + h * DHEAD + tcq,
                    &lQ[r * 128 + tcq]);
  }
#if HAVE_ASYNC_LDS
  wait_async0();
#endif
  __syncthreads();

  // Q A-fragments, kept in registers (4 x K=32 chunks covering D=128)
  v16bf qF[4];
  {
    int row = wave * 16 + l16;
#pragma unroll
    for (int kk = 0; kk < 4; ++kk)
      qF[kk] = lds_frag_a(&lQ[row * 128 + kk * 32], kb8);
  }
  __syncthreads();

  v8f zero = {};
  v8f O[8];
#pragma unroll
  for (int i = 0; i < 8; ++i) O[i] = zero;
  float m_run[8], l_run[8];
#pragma unroll
  for (int r = 0; r < 8; ++r) { m_run[r] = -3.0e38f; l_run[r] = 0.0f; }

  const int qw  = qbase + wave * 16;        // first query row of this wave
  const int nkt = 2 * qb + 2;               // key tiles needed (causal)

  for (int kt = 0; kt < nkt; ++kt) {
    // ---- stage K tile (32x128 bf16): async copy
#pragma unroll
    for (int p = 0; p < 4; ++p) {
      int r = p * 8 + trq;
      size_t srow = (size_t)(kt * 32 + r) * QKV_LD;
      async_copy_b128(qkv + srow + QSIZE + kvh * DHEAD + tcq,
                      &lK[r * 128 + tcq]);
    }
    // ---- stage V^T tile (transpose through VGPRs)
#pragma unroll
    for (int p = 0; p < 4; ++p) {
      int r = p * 8 + trq;
      size_t srow = (size_t)(kt * 32 + r) * QKV_LD;
      v8bf v8 = *(const v8bf*)(qkv + srow + QSIZE + KVSIZE + kvh * DHEAD + tcq);
#pragma unroll
      for (int i = 0; i < 8; ++i) lVt[(tcq + i) * 32 + r] = v8[i];
    }
#if HAVE_ASYNC_LDS
    wait_async0();
#endif
    __syncthreads();

    // skip tiles entirely above the causal diagonal (uniform per wave)
    if (kt * 32 <= qw + 15) {
      // ---- scores: two 16x16 tiles over 32 keys, K-dim = 128
      v8f s0 = zero, s1 = zero;
#pragma unroll
      for (int kk = 0; kk < 4; ++kk) {
        v16bf b0 = lds_frag_b(&lK[(l16)      * 128 + kk * 32], kh16);
        v16bf b1 = lds_frag_b(&lK[(16 + l16) * 128 + kk * 32], kh16);
        s0 = wmma_bf16(qF[kk], b0, s0);
        s1 = wmma_bf16(qF[kk], b1, s1);
      }

      // ---- scale + causal mask
      float x0[8], x1[8];
      const int key0 = kt * 32 + l16;
      const int key1 = kt * 32 + 16 + l16;
#pragma unroll
      for (int r = 0; r < 8; ++r) {
        int qi = qw + r + half8;
        x0[r] = (key0 <= qi) ? s0[r] * scale : -3.0e38f;
        x1[r] = (key1 <= qi) ? s1[r] * scale : -3.0e38f;
      }

      // ---- online softmax (rows live in 16-lane groups: xor 1,2,4,8)
      float m_new[8], corr[8];
#pragma unroll
      for (int r = 0; r < 8; ++r) {
        float t = fmaxf(x0[r], x1[r]);
        t = fmaxf(t, __shfl_xor(t, 1, 32));
        t = fmaxf(t, __shfl_xor(t, 2, 32));
        t = fmaxf(t, __shfl_xor(t, 4, 32));
        t = fmaxf(t, __shfl_xor(t, 8, 32));
        m_new[r] = fmaxf(m_run[r], t);
        corr[r]  = __expf(m_run[r] - m_new[r]);
        m_run[r] = m_new[r];
      }
      float p0[8], p1[8];
#pragma unroll
      for (int r = 0; r < 8; ++r) {
        p0[r] = __expf(x0[r] - m_new[r]);
        p1[r] = __expf(x1[r] - m_new[r]);
        float rs = p0[r] + p1[r];
        rs += __shfl_xor(rs, 1, 32);
        rs += __shfl_xor(rs, 2, 32);
        rs += __shfl_xor(rs, 4, 32);
        rs += __shfl_xor(rs, 8, 32);
        l_run[r] = l_run[r] * corr[r] + rs;
      }
#pragma unroll
      for (int dt = 0; dt < 8; ++dt)
#pragma unroll
        for (int r = 0; r < 8; ++r) O[dt][r] *= corr[r];

      // ---- P: D-layout -> A-layout via wave-private LDS (in-order DS pipe)
      bf16_t* myP = &lP[wave * 512];
#pragma unroll
      for (int r = 0; r < 8; ++r) {
        myP[(r + half8) * 32 + l16]      = (bf16_t)p0[r];
        myP[(r + half8) * 32 + 16 + l16] = (bf16_t)p1[r];
      }
      v16bf pa = lds_frag_a(&myP[l16 * 32], kb8);

      // ---- O += P(16x32) * V(32x128): 8 WMMA over dim tiles
#pragma unroll
      for (int dt = 0; dt < 8; ++dt) {
        v16bf vB = lds_frag_b(&lVt[(dt * 16 + l16) * 32], kh16);
        O[dt] = wmma_bf16(pa, vB, O[dt]);
      }
    }
    __syncthreads();
  }

  // ---- finalize: O / l -> attn[s, h*128 + d] (bf16)
  float inv[8];
#pragma unroll
  for (int r = 0; r < 8; ++r) inv[r] = 1.0f / l_run[r];
#pragma unroll
  for (int dt = 0; dt < 8; ++dt) {
#pragma unroll
    for (int r = 0; r < 8; ++r) {
      int row = qw + r + half8;
      int col = h * DHEAD + dt * 16 + l16;
      attn[(size_t)row * QSIZE + col] = (bf16_t)(O[dt][r] * inv[r]);
    }
  }
}

// ---------------------------------------------------------------------------
// Launch: QKV GEMM (f32->bf16) -> RoPE -> flash attention -> output GEMM
// Workspace: qkv bf16 (2048*6144*2 = 25.2MB) + attn bf16 (2048*4096*2 = 16.8MB)
// ---------------------------------------------------------------------------
extern "C" void kernel_launch(void* const* d_in, const int* in_sizes, int n_in,
                              void* d_out, int out_size, void* d_ws, size_t ws_size,
                              hipStream_t stream) {
  (void)in_sizes; (void)n_in; (void)out_size; (void)ws_size;
  const float* hidden    = (const float*)d_in[0];
  const int*   positions = (const int*)d_in[1];
  const float* Wqkv      = (const float*)d_in[2];
  const float* bqkv      = (const float*)d_in[3];
  const float* Wo        = (const float*)d_in[4];
  float* out = (float*)d_out;

  bf16_t* qkv  = (bf16_t*)d_ws;                    // [2048, 6144] bf16
  bf16_t* attn = qkv + (size_t)SEQ * QKV_LD;       // [2048, 4096] bf16

  // 1) qkv = hidden @ Wqkv^T + bqkv  (A fp32, out bf16)
  dim3 g1(QKV_LD / 128, SEQ / 128);
  gemm_nt<false, true><<<g1, 256, 0, stream>>>(hidden, Wqkv, bqkv, qkv,
                                               SEQ, QKV_LD, HID);
  // 2) RoPE in place on q and k
  int ropeTot = SEQ * (NHEADS + NKVH) * 64;
  rope_kernel<<<(ropeTot + 255) / 256, 256, 0, stream>>>(qkv, positions);

  // 3) causal GQA flash attention (bf16 in/out)
  dim3 g2(SEQ / 64, NHEADS);
  flash_attn<<<g2, 128, 0, stream>>>(qkv, attn);

  // 4) out = attn @ Wo^T  (A bf16, out fp32)
  dim3 g3(HID / 128, SEQ / 128);
  gemm_nt<true, false><<<g3, 256, 0, stream>>>(attn, Wo, nullptr, out,
                                               SEQ, HID, QSIZE);
}
